// MultiLoRALinear_88330297410146
// MI455X (gfx1250) — matrix operand order repaired and checked by reading, not compiled, not run on hardware
//
#include <hip/hip_runtime.h>
#include <hip/hip_bf16.h>
#include <cstdint>

typedef __bf16 bf16_t;
typedef __bf16 v16bf __attribute__((ext_vector_type(16)));
typedef __bf16 v8bf  __attribute__((ext_vector_type(8)));
typedef float  v8f   __attribute__((ext_vector_type(8)));
typedef float  v4f   __attribute__((ext_vector_type(4)));
typedef unsigned int v4u __attribute__((ext_vector_type(4)));
typedef int          v8i __attribute__((ext_vector_type(8)));
typedef int          v4i __attribute__((ext_vector_type(4)));

#define M_TOTAL 16384   // B*S
#define K_DIM   4096    // IN
#define N_DIM   4096    // OUT
#define N_AD    8
#define RANK    64
#define NR      (N_AD * RANK)   // 512

// ---------------------------------------------------------------------------
// Kernel 0: convert x (f32) -> bf16 once. 8 elems / thread, b128 in / b128 out.
// ---------------------------------------------------------------------------
__global__ __launch_bounds__(256) void convert_x_kernel(
    const float* __restrict__ X, bf16_t* __restrict__ Xb)
{
    const size_t i = ((size_t)blockIdx.x * 256 + threadIdx.x) * 8;
    const v4f a = *(const v4f*)(X + i);
    const v4f b = *(const v4f*)(X + i + 4);
    v8bf o;
#pragma unroll
    for (int e = 0; e < 4; ++e) { o[e] = (bf16_t)a[e]; o[4 + e] = (bf16_t)b[e]; }
    *(v8bf*)(Xb + i) = o;
}

// ---------------------------------------------------------------------------
// Kernel 1: fold all adapters into the base weight.
// W_eff[o,i] = W[o,i] + sum_{n,r} scales[n]*mags[n,o]*ups[n,o,r]*downs[n,r,i]
// ---------------------------------------------------------------------------
#define TO 8
__global__ __launch_bounds__(256) void fuse_weights_kernel(
    const float* __restrict__ W, const float* __restrict__ downs,
    const float* __restrict__ ups, const float* __restrict__ mags,
    const float* __restrict__ scales, bf16_t* __restrict__ weff)
{
    __shared__ float coef[TO * NR];
    const int o0  = blockIdx.x * TO;
    const int tid = threadIdx.x;

    for (int idx = tid; idx < TO * NR; idx += 256) {
        const int ol = idx / NR;
        const int nr = idx % NR;
        const int n  = nr >> 6;
        const int r  = nr & 63;
        const int o  = o0 + ol;
        coef[idx] = scales[n] * mags[n * N_DIM + o] * ups[((size_t)n * N_DIM + o) * RANK + r];
    }
    __syncthreads();

    for (int i = tid; i < K_DIM; i += 256) {
        float acc[TO];
#pragma unroll
        for (int ol = 0; ol < TO; ++ol)
            acc[ol] = W[(size_t)(o0 + ol) * K_DIM + i];
        for (int nr = 0; nr < NR; ++nr) {
            const float d = downs[(size_t)nr * K_DIM + i];
#pragma unroll
            for (int ol = 0; ol < TO; ++ol)
                acc[ol] += coef[ol * NR + nr] * d;
        }
#pragma unroll
        for (int ol = 0; ol < TO; ++ol)
            weff[(size_t)(o0 + ol) * K_DIM + i] = (bf16_t)acc[ol];
    }
}

// ---------------------------------------------------------------------------
// Kernel 2: out[m,n] = sum_k xb[m,k] * W_eff[n,k] + b[n]  (bf16 WMMA, f32 acc)
// Block tile 128x128, K-tile 64, LDS double-buffered, tiles staged by the TDM.
// ---------------------------------------------------------------------------
#define BM 128
#define BN 128
#define BK 64
#define LSTR 80  // LDS row stride in bf16 (64 data + 16 pad = 128B + 32B)

union FragAB { v16bf v; v8bf h[2]; };
union FragC  { v8f  v; float f[8]; };

// Issue one TDM 2D tile load: 128 rows x 64 bf16, row stride K_DIM, with LDS
// padding of 8 DWORDs after every 32 DWORDs (=> 160B padded rows = LSTR).
__device__ __forceinline__ void tdm_load_tile(uint32_t lds_addr, uint64_t gaddr,
                                              uint32_t tensor_rows)
{
    v4u g0;
    g0[0] = 1u;                                    // count=1 (valid user D#)
    g0[1] = lds_addr;                              // LDS byte address
    g0[2] = (uint32_t)gaddr;                       // global_addr[31:0]
    g0[3] = (uint32_t)(gaddr >> 32) | (2u << 30);  // global_addr[56:32] | type=2
    v8i g1;
    g1[0] = (int)((1u << 16)     // data_size = 2 bytes
                | (1u << 20)     // pad_enable
                | (4u << 22)     // pad_interval: 32 DWORDs (one 128B row)
                | (7u << 25));   // pad_amount:   8 DWORDs (32B pad)
    g1[1] = (int)((K_DIM & 0xFFFFu) << 16);                        // tensor_dim0 lo16
    g1[2] = (int)((K_DIM >> 16) | ((tensor_rows & 0xFFFFu) << 16)); // dim0 hi | dim1 lo
    g1[3] = (int)((tensor_rows >> 16) | (64u << 16));               // dim1 hi | tile_dim0=64
    g1[4] = 128;                                                    // tile_dim1=128
    g1[5] = K_DIM;                                                  // dim0_stride lo32
    g1[6] = 0;
    g1[7] = 0;
    v4i g2 = {0, 0, 0, 0};
    v4i g3 = {0, 0, 0, 0};
    v8i g4 = {0, 0, 0, 0, 0, 0, 0, 0};
    __builtin_amdgcn_tensor_load_to_lds(g0, g1, g2, g3, g4, 0);
}

__device__ __forceinline__ void compute_stage(
    const bf16_t* tA, const bf16_t* tB, FragC (&acc)[4][2],
    int wm, int wn, int l15, int lhalf)
{
#pragma unroll
    for (int ks = 0; ks < BK; ks += 32) {
        FragAB a[4], b[2];
        // A frag (16x32): lane<16 -> row=lane, K {0..7,16..23};
        //                 lane>=16 -> row=lane-16, K {8..15,24..31}.
#pragma unroll
        for (int tm = 0; tm < 4; ++tm) {
            const bf16_t* pa = tA + (wm * 64 + tm * 16 + l15) * LSTR + ks + lhalf * 8;
            a[tm].h[0] = *(const v8bf*)pa;
            a[tm].h[1] = *(const v8bf*)(pa + 16);
        }
        // B frag (32x16): lane&15 = col n, lane half selects K {0..7,16..23} / {8..15,24..31}.
#pragma unroll
        for (int tn = 0; tn < 2; ++tn) {
            const bf16_t* pb = tB + (wn * 32 + tn * 16 + l15) * LSTR + ks + lhalf * 8;
            b[tn].h[0] = *(const v8bf*)pb;
            b[tn].h[1] = *(const v8bf*)(pb + 16);
        }
#pragma unroll
        for (int tm = 0; tm < 4; ++tm)
#pragma unroll
            for (int tn = 0; tn < 2; ++tn)
                acc[tm][tn].v = __builtin_amdgcn_wmma_f32_16x16x32_bf16(
                    false, a[tm].v, false, b[tn].v,
                    (short)0, acc[tm][tn].v, false, false);
    }
}

__global__ __launch_bounds__(256) void lora_gemm_kernel(
    const bf16_t* __restrict__ Xb, const bf16_t* __restrict__ Weff,
    const float* __restrict__ bias, float* __restrict__ Out)
{
    __shared__ bf16_t lA[2][BM * LSTR];
    __shared__ bf16_t lB[2][BN * LSTR];

    const int tid   = threadIdx.x;
    const int lane  = tid & 31;
    const int wave  = tid >> 5;
    const int wm    = wave >> 2;    // 0..1 -> 64 rows
    const int wn    = wave & 3;     // 0..3 -> 32 cols
    const int m0    = blockIdx.y * BM;
    const int n0    = blockIdx.x * BN;
    const int l15   = lane & 15;
    const int lhalf = lane >> 4;

    const uint32_t ldsA0 = (uint32_t)(uintptr_t)&lA[0][0];
    const uint32_t ldsA1 = (uint32_t)(uintptr_t)&lA[1][0];
    const uint32_t ldsB0 = (uint32_t)(uintptr_t)&lB[0][0];
    const uint32_t ldsB1 = (uint32_t)(uintptr_t)&lB[1][0];
    const uint64_t gA = (uint64_t)(uintptr_t)Xb   + (uint64_t)m0 * K_DIM * 2;
    const uint64_t gB = (uint64_t)(uintptr_t)Weff + (uint64_t)n0 * K_DIM * 2;

    FragC acc[4][2];
#pragma unroll
    for (int i = 0; i < 4; ++i)
#pragma unroll
        for (int j = 0; j < 2; ++j)
#pragma unroll
            for (int k = 0; k < 8; ++k) acc[i][j].f[k] = 0.0f;

    // Prologue: TDM fills buffer 0 with the k0=0 tiles (wave 0 drives the DMA).
    if (wave == 0) {
        tdm_load_tile(ldsA0, gA, (uint32_t)M_TOTAL);
        tdm_load_tile(ldsB0, gB, (uint32_t)N_DIM);
    }

    for (int k0 = 0; k0 < K_DIM; k0 += 2 * BK) {
        // ----- stage A: compute from buffer 0, TDM refills buffer 1 -----
        if (wave == 0) __builtin_amdgcn_s_wait_tensorcnt(0);
        __syncthreads();   // publishes buffer 0 to all waves
        if (wave == 0) {
            tdm_load_tile(ldsA1, gA + (uint64_t)(k0 + BK) * 2, (uint32_t)M_TOTAL);
            tdm_load_tile(ldsB1, gB + (uint64_t)(k0 + BK) * 2, (uint32_t)N_DIM);
        }
        compute_stage(&lA[0][0], &lB[0][0], acc, wm, wn, l15, lhalf);
        __syncthreads();   // all reads of buffer 0 done before its next refill

        // ----- stage B: compute from buffer 1, TDM refills buffer 0 -----
        if (wave == 0) __builtin_amdgcn_s_wait_tensorcnt(0);
        __syncthreads();
        if (wave == 0 && k0 + 2 * BK < K_DIM) {
            tdm_load_tile(ldsA0, gA + (uint64_t)(k0 + 2 * BK) * 2, (uint32_t)M_TOTAL);
            tdm_load_tile(ldsB0, gB + (uint64_t)(k0 + 2 * BK) * 2, (uint32_t)N_DIM);
        }
        compute_stage(&lA[1][0], &lB[1][0], acc, wm, wn, l15, lhalf);
        __syncthreads();
    }

    // Epilogue: C/D layout -> VGPR r, lanes0-15: M=r, N=lane; lanes16-31: M=8+r.
#pragma unroll
    for (int tn = 0; tn < 2; ++tn) {
        const int n  = n0 + wn * 32 + tn * 16 + l15;
        const float bv = bias[n];
#pragma unroll
        for (int tm = 0; tm < 4; ++tm) {
            const int mbase = m0 + wm * 64 + tm * 16 + lhalf * 8;
#pragma unroll
            for (int r = 0; r < 8; ++r)
                Out[(size_t)(mbase + r) * N_DIM + n] = acc[tm][tn].f[r] + bv;
        }
    }
}

// ---------------------------------------------------------------------------
extern "C" void kernel_launch(void* const* d_in, const int* in_sizes, int n_in,
                              void* d_out, int out_size, void* d_ws, size_t ws_size,
                              hipStream_t stream) {
    const float* x      = (const float*)d_in[0];
    const float* W      = (const float*)d_in[1];
    const float* b      = (const float*)d_in[2];
    const float* downs  = (const float*)d_in[3];
    const float* ups    = (const float*)d_in[4];
    const float* mags   = (const float*)d_in[5];
    const float* scales = (const float*)d_in[6];
    float* out = (float*)d_out;

    // Workspace layout: [ x_bf16 : 134 MB ][ W_eff bf16 : 33.5 MB ]
    bf16_t* xb   = (bf16_t*)d_ws;
    bf16_t* weff = (bf16_t*)((char*)d_ws + (size_t)M_TOTAL * K_DIM * sizeof(bf16_t));

    convert_x_kernel<<<(M_TOTAL * (size_t)K_DIM) / (256 * 8), 256, 0, stream>>>(x, xb);
    fuse_weights_kernel<<<N_DIM / TO, 256, 0, stream>>>(W, downs, ups, mags, scales, weff);

    dim3 grid(N_DIM / BN, M_TOTAL / BM);  // x = n (fastest) for L2 stripe reuse
    lora_gemm_kernel<<<grid, 256, 0, stream>>>(xb, weff, b, out);
}
